// MultiHeadAttention_1090921693387
// MI455X (gfx1250) — matrix operand order repaired and checked by reading, not compiled, not run on hardware
//
#include <hip/hip_runtime.h>
#include <stdint.h>

#define D_MODEL 1024
#define S_LEN   2048
#define N_HEADS 16
#define D_K     64
#define BATCH   4

typedef __bf16 bf16_t;
typedef __attribute__((ext_vector_type(16))) __bf16 v16bf;
typedef __attribute__((ext_vector_type(8)))  float  v8f;
typedef unsigned int u32x4 __attribute__((ext_vector_type(4)));
typedef int          i32x4 __attribute__((ext_vector_type(4)));
typedef int          i32x8 __attribute__((ext_vector_type(8)));

union FragU { v16bf v; uint32_t u[8]; };

// ---------------------------------------------------------------------------
// Fragment builders from LDS for CDNA5 WMMA 16x16x32 bf16 (A/B share a contiguous k order)
// (CDNA5 ISA §7.12.2).
// A (16x32, MxK): lane L -> m = L%16, half = L/16.
//   elems = k in [half*16, +16)
// B (32x16, KxN): lane L -> n = L%16, half = L/16; elems = k in [half*16,+16)
// C/D (16x16 f32): lane L -> n = L%16; VGPR r -> m = r + 8*(L/16).
// ---------------------------------------------------------------------------
__device__ inline v16bf lds_load_a(const bf16_t* base, int ld, int mBase, int kBase) {
  const int lane = threadIdx.x & 31;
  const int m    = mBase + (lane & 15);
  const int half = lane >> 4;
  const uint32_t* p0 = (const uint32_t*)(base + m * ld + kBase + half * 16);
  const uint32_t* p1 = (const uint32_t*)(base + m * ld + kBase + half * 16 + 8);
  FragU f;
  f.u[0] = p0[0]; f.u[1] = p0[1]; f.u[2] = p0[2]; f.u[3] = p0[3];
  f.u[4] = p1[0]; f.u[5] = p1[1]; f.u[6] = p1[2]; f.u[7] = p1[3];
  return f.v;
}

__device__ inline v16bf lds_load_b(const bf16_t* base, int ld, int nBase, int kBase) {
  const int lane = threadIdx.x & 31;
  const int n    = nBase + (lane & 15);
  const int half = lane >> 4;
  const uint32_t* p = (const uint32_t*)(base + n * ld + kBase + half * 16);
  FragU f;
#pragma unroll
  for (int i = 0; i < 8; ++i) f.u[i] = p[i];
  return f.v;
}

__device__ inline v8f wmma_bf16(v16bf a, v16bf b, v8f c) {
  return __builtin_amdgcn_wmma_f32_16x16x32_bf16(false, a, false, b, (short)0, c,
                                                 false, false);
}

// ---------------------------------------------------------------------------
// TDM: load a 64x64 bf16 2-D tile (row length 64 elems = 128B, row stride
// `stride_elems`) into LDS at byte offset `lds_off`, with hardware padding of
// 4 dwords after every 32 dwords -> LDS row stride 72 bf16 (144B).
// D# packing per CDNA5 ISA (groups 2/3 zero => 2-D).
// Issue from ONE wave only; completion via TENSORcnt.
// This toolchain exposes the 6-arg clang-23 builtin form.
// ---------------------------------------------------------------------------
__device__ inline void tdm_load_tile64(uint32_t lds_off, const bf16_t* gsrc,
                                       uint32_t stride_elems) {
  const uint64_t ga = (uint64_t)(size_t)gsrc;
  u32x4 g0;
  g0[0] = 1u;                                             // count=1 (valid)
  g0[1] = lds_off;                                        // lds_addr
  g0[2] = (uint32_t)ga;                                   // global_addr[31:0]
  g0[3] = (uint32_t)((ga >> 32) & 0x1FFFFFFu) | (2u << 30); // ga[56:32] | type=2
  i32x8 g1;
  g1[0] = (1 << 16)      // data_size = 1 -> 2 bytes
        | (1 << 20)      // pad_enable
        | (4 << 22)      // pad_interval = 4 -> every 32 dwords (one 64-elem row)
        | (3 << 25);     // pad_amount = 3 -> 4 dwords (8 bf16) -> stride 72
  g1[1] = (64 << 16);    // tensor_dim0 = 64
  g1[2] = (64 << 16);    // tensor_dim1[15:0] = 64
  g1[3] = (64 << 16);    // tile_dim0 = 64
  g1[4] = 64;            // tile_dim1 = 64, tile_dim2 = 0
  g1[5] = (int)stride_elems;  // tensor_dim0_stride[31:0]
  g1[6] = 0;
  g1[7] = 0;
  const i32x4 z4 = {0, 0, 0, 0};
  const i32x8 z8 = {0, 0, 0, 0, 0, 0, 0, 0};
  __builtin_amdgcn_tensor_load_to_lds(g0, g1, z4, z4, z8, 0);
}

// ---------------------------------------------------------------------------
// Kernel 1: fused QKV projection.  out = x @ W^T + b   (NT gemm, fp32 in)
// Block computes 128x64 tile of [8192 x 1024]; grid.z in {0,1,2} = Q/K/V.
// Q, K written as bf16 [bh][s][d]; V written transposed bf16 [bh][d][s].
// ---------------------------------------------------------------------------
__global__ __launch_bounds__(256) void qkv_proj_kernel(
    const float* __restrict__ x,
    const float* __restrict__ Wq, const float* __restrict__ bq,
    const float* __restrict__ Wk, const float* __restrict__ bk,
    const float* __restrict__ Wv, const float* __restrict__ bv,
    bf16_t* __restrict__ Qh, bf16_t* __restrict__ Kh, bf16_t* __restrict__ Vth) {
  const int which = blockIdx.z;
  const float* W    = (which == 0) ? Wq : (which == 1) ? Wk : Wv;
  const float* bias = (which == 0) ? bq : (which == 1) ? bk : bv;

  __shared__ bf16_t As[128][40];  // [m][k], row stride 80B (16B aligned)
  __shared__ bf16_t Bs[64][40];   // [n][k] == W row-major slice

  const int tid  = threadIdx.x;
  const int lane = tid & 31;
  const int wid  = tid >> 5;
  const int wm   = (wid & 3) * 32;
  const int wn   = (wid >> 2) * 32;
  const int m0   = blockIdx.x * 128;
  const int n0   = blockIdx.y * 64;

  v8f acc[2][2];
#pragma unroll
  for (int i = 0; i < 2; ++i)
#pragma unroll
    for (int j = 0; j < 2; ++j) acc[i][j] = (v8f){0.f, 0.f, 0.f, 0.f, 0.f, 0.f, 0.f, 0.f};

  const int lr = tid >> 3;        // 0..31
  const int lc = (tid & 7) * 4;   // float4 column

  for (int k0 = 0; k0 < D_MODEL; k0 += 32) {
    if (k0 + 32 < D_MODEL) {  // prefetch next K-slice -> global_prefetch_b8
      __builtin_prefetch(&x[(size_t)(m0 + lr) * D_MODEL + k0 + 32 + lc], 0, 3);
      __builtin_prefetch(&W[(size_t)(n0 + lr) * D_MODEL + k0 + 32 + lc], 0, 3);
    }
    __syncthreads();
#pragma unroll
    for (int rr = 0; rr < 128; rr += 32) {  // stage A (x) 128x32 fp32 -> bf16
      const float4 f = *(const float4*)&x[(size_t)(m0 + lr + rr) * D_MODEL + k0 + lc];
      bf16_t* dst = &As[lr + rr][lc];
      dst[0] = (bf16_t)f.x; dst[1] = (bf16_t)f.y; dst[2] = (bf16_t)f.z; dst[3] = (bf16_t)f.w;
    }
#pragma unroll
    for (int rr = 0; rr < 64; rr += 32) {   // stage B (W rows) 64x32 fp32 -> bf16
      const float4 f = *(const float4*)&W[(size_t)(n0 + lr + rr) * D_MODEL + k0 + lc];
      bf16_t* dst = &Bs[lr + rr][lc];
      dst[0] = (bf16_t)f.x; dst[1] = (bf16_t)f.y; dst[2] = (bf16_t)f.z; dst[3] = (bf16_t)f.w;
    }
    __syncthreads();

    const v16bf a0 = lds_load_a(&As[0][0], 40, wm, 0);
    const v16bf a1 = lds_load_a(&As[0][0], 40, wm + 16, 0);
    const v16bf b0 = lds_load_b(&Bs[0][0], 40, wn, 0);
    const v16bf b1 = lds_load_b(&Bs[0][0], 40, wn + 16, 0);
    acc[0][0] = wmma_bf16(a0, b0, acc[0][0]);
    acc[0][1] = wmma_bf16(a0, b1, acc[0][1]);
    acc[1][0] = wmma_bf16(a1, b0, acc[1][0]);
    acc[1][1] = wmma_bf16(a1, b1, acc[1][1]);
  }

  const int half = lane >> 4;
  const int nloc = lane & 15;
#pragma unroll
  for (int i = 0; i < 2; ++i) {
#pragma unroll
    for (int j = 0; j < 2; ++j) {
      const int n  = n0 + wn + j * 16 + nloc;
      const float bb = bias[n];
      const int h = n >> 6, d = n & 63;
#pragma unroll
      for (int r = 0; r < 8; ++r) {
        const int m = m0 + wm + i * 16 + half * 8 + r;
        const float val = acc[i][j][r] + bb;
        const int b = m >> 11;
        const int s = m & 2047;
        const size_t bh = (size_t)(b * N_HEADS + h);
        if (which == 2) {
          Vth[(bh * D_K + d) * S_LEN + s] = (bf16_t)val;
        } else {
          bf16_t* dst = (which == 0) ? Qh : Kh;
          dst[(bh * S_LEN + s) * D_K + d] = (bf16_t)val;
        }
      }
    }
  }
}

// ---------------------------------------------------------------------------
// Kernel 2: flash attention per (bh, q-block of 64 rows).
// K/V (and Q) tiles streamed by the Tensor Data Mover into double-buffered
// LDS (hardware-padded to stride 72); online softmax; all matmuls WMMA bf16.
// ---------------------------------------------------------------------------
__global__ __launch_bounds__(256) void attn_kernel(
    const bf16_t* __restrict__ Qh, const bf16_t* __restrict__ Kh,
    const bf16_t* __restrict__ Vth, const int* __restrict__ amask,
    bf16_t* __restrict__ attnOut) {
  __shared__ bf16_t Qs[64][72];      // [q][d]
  __shared__ bf16_t Ks[2][64][72];   // [kv][d], ping-pong
  __shared__ bf16_t Vs[2][64][72];   // [d][kv], ping-pong (from V transposed)
  __shared__ bf16_t Ps[64][72];      // probs bf16 [q][kv]
  __shared__ float  Ss[64][68];      // scores fp32 [q][kv]
  __shared__ float  rowM[64], rowL[64], rowF[64];
  __shared__ int    smask[64];

  const int tid  = threadIdx.x;
  const int lane = tid & 31;
  const int wid  = tid >> 5;
  const int q0   = blockIdx.x * 64;
  const int bh   = blockIdx.y;
  const int b    = bh >> 4;
  const int h    = bh & 15;

  const bf16_t* Kbase = Kh + (size_t)bh * S_LEN * D_K;
  const bf16_t* Vbase = Vth + (size_t)bh * D_K * S_LEN;

  // TDM prologue: Q tile + first K/V tiles (wave 0 issues; TENSORcnt tracks)
  if (wid == 0) {
    tdm_load_tile64((uint32_t)(size_t)&Qs[0][0], Qh + ((size_t)bh * S_LEN + q0) * D_K, D_K);
    tdm_load_tile64((uint32_t)(size_t)&Ks[0][0][0], Kbase, D_K);
    tdm_load_tile64((uint32_t)(size_t)&Vs[0][0][0], Vbase, S_LEN);
  }
  if (tid < 64) { rowM[tid] = -3.0e38f; rowL[tid] = 0.f; }

  const int tm  = (wid & 3) * 16;   // wave's q-tile row base
  const int tn2 = (wid >> 2) * 32;  // wave's column pair base (kv for S, d for O)
  v8f oacc[2];
  oacc[0] = (v8f){0.f, 0.f, 0.f, 0.f, 0.f, 0.f, 0.f, 0.f};
  oacc[1] = oacc[0];

  const int half = lane >> 4;
  const int nloc = lane & 15;

  for (int kb = 0; kb < S_LEN / 64; ++kb) {
    const int k0  = kb * 64;
    const int buf = kb & 1;

    if (tid < 64) smask[tid] = amask[(size_t)b * S_LEN + k0 + tid];
    if (wid == 0) __builtin_amdgcn_s_wait_tensorcnt(0);  // tiles for kb landed
    __syncthreads();  // release: Qs/Ks[buf]/Vs[buf]/smask visible to all waves

    // Prefetch next K/V tiles into the other buffer (its readers finished
    // in iteration kb-1, before the barrier above).
    if (wid == 0 && kb + 1 < S_LEN / 64) {
      tdm_load_tile64((uint32_t)(size_t)&Ks[buf ^ 1][0][0],
                      Kbase + (size_t)(k0 + 64) * D_K, D_K);
      tdm_load_tile64((uint32_t)(size_t)&Vs[buf ^ 1][0][0],
                      Vbase + (size_t)(k0 + 64), S_LEN);
    }

    // ---- scores: S = (Q K^T) / sqrt(d_k); 2 tiles per wave ----
#pragma unroll
    for (int j = 0; j < 2; ++j) {
      const int tn = tn2 + j * 16;
      v8f sacc = (v8f){0.f, 0.f, 0.f, 0.f, 0.f, 0.f, 0.f, 0.f};
#pragma unroll
      for (int kc = 0; kc < 2; ++kc) {
        const v16bf a  = lds_load_a(&Qs[0][0], 72, tm, kc * 32);
        const v16bf bb = lds_load_b(&Ks[buf][0][0], 72, tn, kc * 32);
        sacc = wmma_bf16(a, bb, sacc);
      }
      const int nn = tn + nloc;
#pragma unroll
      for (int r = 0; r < 8; ++r)
        Ss[tm + half * 8 + r][nn] = sacc[r] * 0.125f;  // 1/sqrt(64)
    }
    __syncthreads();

    // ---- online softmax bookkeeping: one thread per q row ----
    if (tid < 64) {
      const int r = tid;
      const float mold = rowM[r];
      float rmax = -__builtin_inff();
      for (int j = 0; j < 64; ++j) {
        const float s = smask[j] ? Ss[r][j] : -__builtin_inff();
        rmax = fmaxf(rmax, s);
      }
      const float mnew = fmaxf(mold, rmax);  // mold finite => mnew finite
      const float f    = __expf(mold - mnew);
      float lsum = 0.f;
      for (int j = 0; j < 64; ++j) {
        const float s = smask[j] ? Ss[r][j] : -__builtin_inff();
        const float p = __expf(s - mnew);
        lsum += p;
        Ps[r][j] = (bf16_t)p;
      }
      rowF[r] = f;
      rowM[r] = mnew;
      rowL[r] = rowL[r] * f + lsum;
    }
    __syncthreads();

    // ---- O = O*f + P @ V ----
#pragma unroll
    for (int j = 0; j < 2; ++j) {
      const int td = tn2 + j * 16;
#pragma unroll
      for (int r = 0; r < 8; ++r) oacc[j][r] *= rowF[tm + half * 8 + r];
#pragma unroll
      for (int kc = 0; kc < 2; ++kc) {
        const v16bf a  = lds_load_a(&Ps[0][0], 72, tm, kc * 32);
        const v16bf bb = lds_load_b(&Vs[buf][0][0], 72, td, kc * 32);
        oacc[j] = wmma_bf16(a, bb, oacc[j]);
      }
    }
  }

  __syncthreads();
#pragma unroll
  for (int j = 0; j < 2; ++j) {
    const int dcol = tn2 + j * 16 + nloc;
#pragma unroll
    for (int r = 0; r < 8; ++r) {
      const int m = tm + half * 8 + r;
      const float l   = rowL[m];
      const float inv = (l > 0.f) ? (1.0f / l) : 0.f;
      const size_t row = (size_t)b * S_LEN + q0 + m;
      attnOut[row * D_MODEL + h * D_K + dcol] = (bf16_t)(oacc[j][r] * inv);
    }
  }
}

// ---------------------------------------------------------------------------
// Kernel 3: output projection.  out = attn @ Wo^T + bo  (bf16 A, fp32 out)
// ---------------------------------------------------------------------------
__global__ __launch_bounds__(256) void out_proj_kernel(
    const bf16_t* __restrict__ attn, const float* __restrict__ Wo,
    const float* __restrict__ bo, float* __restrict__ out) {
  __shared__ bf16_t As[128][40];
  __shared__ bf16_t Bs[64][40];

  const int tid  = threadIdx.x;
  const int lane = tid & 31;
  const int wid  = tid >> 5;
  const int wm   = (wid & 3) * 32;
  const int wn   = (wid >> 2) * 32;
  const int m0   = blockIdx.x * 128;
  const int n0   = blockIdx.y * 64;

  v8f acc[2][2];
#pragma unroll
  for (int i = 0; i < 2; ++i)
#pragma unroll
    for (int j = 0; j < 2; ++j) acc[i][j] = (v8f){0.f, 0.f, 0.f, 0.f, 0.f, 0.f, 0.f, 0.f};

  const int ar = tid >> 1;          // 0..127
  const int ac = (tid & 1) * 8;     // dword column (8 dwords = 16 bf16)
  const int br = tid >> 3;          // 0..31
  const int bc = (tid & 7) * 4;

  for (int k0 = 0; k0 < D_MODEL; k0 += 32) {
    if (k0 + 32 < D_MODEL) {
      __builtin_prefetch(&attn[(size_t)(m0 + ar) * D_MODEL + k0 + 32], 0, 3);
      __builtin_prefetch(&Wo[(size_t)(n0 + br) * D_MODEL + k0 + 32 + bc], 0, 3);
    }
    __syncthreads();
    {
      const uint32_t* src = (const uint32_t*)&attn[(size_t)(m0 + ar) * D_MODEL + k0] + ac;
      uint32_t* dst = (uint32_t*)&As[ar][ac * 2];
#pragma unroll
      for (int i = 0; i < 8; ++i) dst[i] = src[i];
    }
#pragma unroll
    for (int rr = 0; rr < 64; rr += 32) {
      const float4 f = *(const float4*)&Wo[(size_t)(n0 + br + rr) * D_MODEL + k0 + bc];
      bf16_t* dst = &Bs[br + rr][bc];
      dst[0] = (bf16_t)f.x; dst[1] = (bf16_t)f.y; dst[2] = (bf16_t)f.z; dst[3] = (bf16_t)f.w;
    }
    __syncthreads();

    const v16bf a0 = lds_load_a(&As[0][0], 40, wm, 0);
    const v16bf a1 = lds_load_a(&As[0][0], 40, wm + 16, 0);
    const v16bf b0 = lds_load_b(&Bs[0][0], 40, wn, 0);
    const v16bf b1 = lds_load_b(&Bs[0][0], 40, wn + 16, 0);
    acc[0][0] = wmma_bf16(a0, b0, acc[0][0]);
    acc[0][1] = wmma_bf16(a0, b1, acc[0][1]);
    acc[1][0] = wmma_bf16(a1, b0, acc[1][0]);
    acc[1][1] = wmma_bf16(a1, b1, acc[1][1]);
  }

  const int half = lane >> 4;
  const int nloc = lane & 15;
#pragma unroll
  for (int i = 0; i < 2; ++i) {
#pragma unroll
    for (int j = 0; j < 2; ++j) {
      const int n = n0 + wn + j * 16 + nloc;
      const float bb = bo[n];
#pragma unroll
      for (int r = 0; r < 8; ++r) {
        const int m = m0 + wm + i * 16 + half * 8 + r;
        out[(size_t)m * D_MODEL + n] = acc[i][j][r] + bb;
      }
    }
  }
}

// ---------------------------------------------------------------------------
extern "C" void kernel_launch(void* const* d_in, const int* in_sizes, int n_in,
                              void* d_out, int out_size, void* d_ws, size_t ws_size,
                              hipStream_t stream) {
  (void)in_sizes; (void)n_in; (void)out_size; (void)ws_size;
  const float* x  = (const float*)d_in[0];
  const int* am   = (const int*)d_in[1];
  const float* Wq = (const float*)d_in[2];
  const float* bq = (const float*)d_in[3];
  const float* Wk = (const float*)d_in[4];
  const float* bk = (const float*)d_in[5];
  const float* Wv = (const float*)d_in[6];
  const float* bv = (const float*)d_in[7];
  const float* Wo = (const float*)d_in[8];
  const float* bo = (const float*)d_in[9];
  float* out = (float*)d_out;

  char* ws = (char*)d_ws;
  const size_t QKV_BYTES = (size_t)BATCH * N_HEADS * S_LEN * D_K * sizeof(bf16_t); // 16 MB
  bf16_t* Qh   = (bf16_t*)(ws);
  bf16_t* Kh   = (bf16_t*)(ws + QKV_BYTES);
  bf16_t* Vth  = (bf16_t*)(ws + 2 * QKV_BYTES);
  bf16_t* attn = (bf16_t*)(ws + 3 * QKV_BYTES);

  qkv_proj_kernel<<<dim3(64, 16, 3), 256, 0, stream>>>(x, Wq, bq, Wk, bk, Wv, bv,
                                                       Qh, Kh, Vth);
  attn_kernel<<<dim3(S_LEN / 64, BATCH * N_HEADS), 256, 0, stream>>>(Qh, Kh, Vth, am,
                                                                     attn);
  out_proj_kernel<<<dim3(64, 16), 256, 0, stream>>>(attn, Wo, bo, out);
}
